// SpikeClassifier_6296422056534
// MI455X (gfx1250) — matrix operand order repaired and checked by reading, not compiled, not run on hardware
//
#include <hip/hip_runtime.h>
#include <hip/hip_bf16.h>
#include <math.h>

typedef __bf16 bf16;
typedef __attribute__((ext_vector_type(16))) __bf16 v16bf;
typedef __attribute__((ext_vector_type(8)))  float  v8f;

#define TT 4
#define DD 128   // input feature dim
#define HH 256   // hidden / embed dim
#define CC 40    // classes
#define CP 48    // classes padded to 3x16 tiles

// ---------------------------------------------------------------- WMMA core
__device__ __forceinline__ v8f wmma_bf16(v16bf a, v16bf b, v8f c) {
  return __builtin_amdgcn_wmma_f32_16x16x32_bf16(
      /*neg_a=*/false, a, /*neg_b=*/false, b,
      /*c_mod=*/(short)0, c, /*reuse_a=*/false, /*reuse_b=*/false);
}

// A fragment (16x32 bf16): lane holds row (lane&15); K groups: k0..k0+7 and k0+16..k0+23
__device__ __forceinline__ v16bf frag_a_f32(const float* __restrict__ rowp, int k0) {
  v16bf a;
#pragma unroll
  for (int i = 0; i < 8; ++i) a[i] = (bf16)rowp[k0 + i];
#pragma unroll
  for (int i = 0; i < 8; ++i) a[8 + i] = (bf16)rowp[k0 + 16 + i];
  return a;
}
__device__ __forceinline__ v16bf frag_a_bf16(const bf16* __restrict__ rowp, int k0) {
  v16bf a;
#pragma unroll
  for (int i = 0; i < 8; ++i) a[i] = rowp[k0 + i];
#pragma unroll
  for (int i = 0; i < 8; ++i) a[8 + i] = rowp[k0 + 16 + i];
  return a;
}
// B fragment (32x16 bf16): lane holds col (lane&15); K groups: k0..k0+7 and k0+16..k0+23
// colp points at column-major (transposed) weights: Wt[col*K + k]
__device__ __forceinline__ v16bf frag_b(const bf16* __restrict__ colp, int k0) {
  v16bf b;
#pragma unroll
  for (int i = 0; i < 8; ++i) b[i] = colp[k0 + i];
#pragma unroll
  for (int i = 0; i < 8; ++i) b[8 + i] = colp[k0 + 16 + i];
  return b;
}

// ------------------------------------------------------------ small kernels
__global__ void k_deg_init(float* deg, int N) {
  int i = blockIdx.x * blockDim.x + threadIdx.x;
  if (i < N) deg[i] = 1.0f;  // self loop
}
__global__ void k_deg_count(const int* __restrict__ ei, float* deg, int E) {
  int e = blockIdx.x * blockDim.x + threadIdx.x;
  if (e < E) atomicAdd(&deg[ei[E + e]], 1.0f);  // dst = ei[1][e]
}
__global__ void k_dinv(const float* __restrict__ deg, float* dinv, int N) {
  int i = blockIdx.x * blockDim.x + threadIdx.x;
  if (i < N) dinv[i] = rsqrtf(fmaxf(deg[i], 1.0f));
}
// Transpose + pad weights to bf16 column-major: Wt[c*K + k] = W[k*Cin + c] (0 if c>=Cin)
__global__ void k_prep_w(const float* __restrict__ W, bf16* Wt, int K, int Cin, int total) {
  int idx = blockIdx.x * blockDim.x + threadIdx.x;
  if (idx >= total) return;
  int c = idx / K, k = idx - c * K;
  Wt[idx] = (c < Cin) ? (bf16)W[(size_t)k * Cin + c] : (bf16)0.0f;
}
// agg1[n,f] = feat[n,f] * dinv[n]^2   (self-loop term; also serves as init)
__global__ void k_init_agg1(const float* __restrict__ feat, const float* __restrict__ dinv,
                            float* agg1, int N) {
  int gid = blockIdx.x * blockDim.x + threadIdx.x;
  if (gid >= N * DD) return;
  int n = gid >> 7;
  float w = dinv[n];
  agg1[gid] = feat[gid] * w * w;
}
// one wave per edge; 32 lanes x 4 features
__global__ void k_edge_agg1(const int* __restrict__ ei, const float* __restrict__ feat,
                            const float* __restrict__ dinv, float* agg1, int E) {
  int tid = blockIdx.x * blockDim.x + threadIdx.x;
  int wid = tid >> 5, lane = tid & 31;
  if (wid >= E) return;
  int s = ei[wid], d = ei[E + wid];
  float w = dinv[s] * dinv[d];
  const float* fs = feat + (size_t)s * DD;
  float* ad = agg1 + (size_t)d * DD;
#pragma unroll
  for (int j = 0; j < 4; ++j) {
    int f = lane + (j << 5);
    atomicAdd(ad + f, fs[f] * w);
  }
}
// agg2[t,n,f] = s1[t,n,f] * dinv[n]^2  (self-loop + init), flat over T*N*HH
__global__ void k_init_agg2(const bf16* __restrict__ s1, const float* __restrict__ dinv,
                            float* agg2, int N, int total) {
  int gid = blockIdx.x * blockDim.x + threadIdx.x;
  if (gid >= total) return;
  int node = (gid >> 8) % N;
  float w = dinv[node];
  agg2[gid] = (float)s1[gid] * w * w;
}
// one wave per (edge, t); 32 lanes x 8 features
__global__ void k_edge_agg2(const int* __restrict__ ei, const bf16* __restrict__ s1,
                            const float* __restrict__ dinv, float* agg2, int E, int N) {
  int tid = blockIdx.x * blockDim.x + threadIdx.x;
  int wid = tid >> 5, lane = tid & 31;
  if (wid >= E * TT) return;
  int e = wid >> 2, t = wid & 3;
  int s = ei[e], d = ei[E + e];
  float w = dinv[s] * dinv[d];
  const bf16* ss = s1 + ((size_t)t * N + s) * HH;
  float* ad = agg2 + ((size_t)t * N + d) * HH;
#pragma unroll
  for (int j = 0; j < 8; ++j) {
    int f = lane + (j << 5);
    atomicAdd(ad + f, (float)ss[f] * w);
  }
}

// -------------------- GEMM1 (K=128) + LIF; 16x64 tile/wave (4 N-subtiles), z const over t
__global__ void k_gemm1_lif(const float* __restrict__ agg1, const bf16* __restrict__ W1t,
                            const float* __restrict__ b1, bf16* s1, int N) {
  int tid = blockIdx.x * blockDim.x + threadIdx.x;
  int wid = tid >> 5, lane = tid & 31;
  int mtiles = N >> 4;
  if (wid >= mtiles * 4) return;   // wave-uniform (16 ntiles / 4 per wave)
  int mt = wid >> 2, ntb = wid & 3;
  int row = (mt << 4) + (lane & 15);
  int col0 = (ntb << 6) + (lane & 15);   // + q*16
  int hk = lane >> 4;
  const float* arow = agg1 + (size_t)row * DD;
  const bf16* bcol = W1t + (size_t)col0 * DD;
  v8f c[4];
#pragma unroll
  for (int q = 0; q < 4; ++q) c[q] = (v8f){};
#pragma unroll
  for (int kc = 0; kc < DD / 32; ++kc) {
    v16bf a = frag_a_f32(arow, kc * 32 + (hk << 3));   // load A once
#pragma unroll
    for (int q = 0; q < 4; ++q) {                      // 4 WMMAs reuse it
      v16bf b = frag_b(bcol + (size_t)q * 16 * DD, kc * 32 + (hk << 3));
      c[q] = wmma_bf16(a, b, c[q]);
    }
  }
  int rbase = (mt << 4) + (hk << 3);
  float v[4][8];
#pragma unroll
  for (int q = 0; q < 4; ++q)
#pragma unroll
    for (int r = 0; r < 8; ++r) v[q][r] = 0.0f;
#pragma unroll
  for (int t = 0; t < TT; ++t) {
#pragma unroll
    for (int q = 0; q < 4; ++q) {
      int col = col0 + (q << 4);
      float bias = b1[col];
#pragma unroll
      for (int r = 0; r < 8; ++r) {
        float z = c[q][r] + bias;
        v[q][r] = 0.5f * (v[q][r] + z);                  // v += (x - v)/tau, tau=2
        float sp = (v[q][r] >= 1.0f) ? 1.0f : 0.0f;      // H(v - vth)
        v[q][r] *= (1.0f - sp);                          // hard reset
        s1[((size_t)t * N + rbase + r) * HH + col] = (bf16)sp;
      }
    }
  }
}

// -------------------- GEMM2 (K=256) + sequential-t LIF; 16x64 tile/wave
__global__ void k_gemm2_lif(const float* __restrict__ agg2, const bf16* __restrict__ W2t,
                            const float* __restrict__ b2, bf16* s2, int N) {
  int tid = blockIdx.x * blockDim.x + threadIdx.x;
  int wid = tid >> 5, lane = tid & 31;
  int mtiles = N >> 4;
  if (wid >= mtiles * 4) return;
  int mt = wid >> 2, ntb = wid & 3;
  int row = (mt << 4) + (lane & 15);
  int col0 = (ntb << 6) + (lane & 15);
  int hk = lane >> 4;
  const bf16* bcol = W2t + (size_t)col0 * HH;
  int rbase = (mt << 4) + (hk << 3);
  float v[4][8];
#pragma unroll
  for (int q = 0; q < 4; ++q)
#pragma unroll
    for (int r = 0; r < 8; ++r) v[q][r] = 0.0f;
#pragma unroll
  for (int t = 0; t < TT; ++t) {
    const float* arow = agg2 + ((size_t)t * N + row) * HH;
    v8f c[4];
#pragma unroll
    for (int q = 0; q < 4; ++q) c[q] = (v8f){};
#pragma unroll
    for (int kc = 0; kc < HH / 32; ++kc) {
      v16bf a = frag_a_f32(arow, kc * 32 + (hk << 3));   // A read once per kc
#pragma unroll
      for (int q = 0; q < 4; ++q) {
        v16bf b = frag_b(bcol + (size_t)q * 16 * HH, kc * 32 + (hk << 3));
        c[q] = wmma_bf16(a, b, c[q]);
      }
    }
#pragma unroll
    for (int q = 0; q < 4; ++q) {
      int col = col0 + (q << 4);
      float bias = b2[col];
#pragma unroll
      for (int r = 0; r < 8; ++r) {
        float z = c[q][r] + bias;
        v[q][r] = 0.5f * (v[q][r] + z);
        float sp = (v[q][r] >= 1.0f) ? 1.0f : 0.0f;
        v[q][r] *= (1.0f - sp);
        s2[((size_t)t * N + rbase + r) * HH + col] = (bf16)sp;
      }
    }
  }
}

// -------------------- FC (K=256, 48 padded cols in one wave) + LIF + log-mean
__global__ void k_fc_lif_out(const bf16* __restrict__ s2, const bf16* __restrict__ fcWt,
                             const float* __restrict__ fcb, float* out, int N) {
  int tid = blockIdx.x * blockDim.x + threadIdx.x;
  int wid = tid >> 5, lane = tid & 31;
  int mtiles = N >> 4;
  if (wid >= mtiles) return;   // wave-uniform; one wave = 16 rows x all 48 cols
  int mt = wid;
  int row = (mt << 4) + (lane & 15);
  int col0 = (lane & 15);      // + q*16, q=0..2
  int hk = lane >> 4;
  const bf16* bcol = fcWt + (size_t)col0 * HH;
  int rbase = (mt << 4) + (hk << 3);
  float v[3][8], ssum[3][8];
#pragma unroll
  for (int q = 0; q < 3; ++q)
#pragma unroll
    for (int r = 0; r < 8; ++r) { v[q][r] = 0.0f; ssum[q][r] = 0.0f; }
#pragma unroll
  for (int t = 0; t < TT; ++t) {
    const bf16* arow = s2 + ((size_t)t * N + row) * HH;
    v8f c[3];
#pragma unroll
    for (int q = 0; q < 3; ++q) c[q] = (v8f){};
#pragma unroll
    for (int kc = 0; kc < HH / 32; ++kc) {
      v16bf a = frag_a_bf16(arow, kc * 32 + (hk << 3));  // spike tensor read once
#pragma unroll
      for (int q = 0; q < 3; ++q) {
        v16bf b = frag_b(bcol + (size_t)q * 16 * HH, kc * 32 + (hk << 3));
        c[q] = wmma_bf16(a, b, c[q]);
      }
    }
#pragma unroll
    for (int q = 0; q < 3; ++q) {
      int col = col0 + (q << 4);
      float bias = (col < CC) ? fcb[col] : 0.0f;
#pragma unroll
      for (int r = 0; r < 8; ++r) {
        float z = c[q][r] + bias;
        v[q][r] = 0.5f * (v[q][r] + z);
        float sp = (v[q][r] >= 1.0f) ? 1.0f : 0.0f;
        v[q][r] *= (1.0f - sp);
        ssum[q][r] += sp;
      }
    }
  }
#pragma unroll
  for (int q = 0; q < 3; ++q) {
    int col = col0 + (q << 4);
    if (col < CC) {
#pragma unroll
      for (int r = 0; r < 8; ++r)
        out[(size_t)(rbase + r) * CC + col] = logf(ssum[q][r] * 0.25f + 1e-6f);
    }
  }
}

// ------------------------------------------------------------------- launch
extern "C" void kernel_launch(void* const* d_in, const int* in_sizes, int n_in,
                              void* d_out, int out_size, void* d_ws, size_t ws_size,
                              hipStream_t stream) {
  const float* feat = (const float*)d_in[0];
  const int*   ei   = (const int*)d_in[1];
  const float* W1   = (const float*)d_in[2];
  const float* b1   = (const float*)d_in[3];
  const float* W2   = (const float*)d_in[4];
  const float* b2   = (const float*)d_in[5];
  const float* fcW  = (const float*)d_in[6];
  const float* fcb  = (const float*)d_in[7];
  float* out = (float*)d_out;

  const int N = in_sizes[0] / DD;   // 100000 (multiple of 16)
  const int E = in_sizes[1] / 2;    // 1600000

  // workspace carve (256B aligned)
  char* ws = (char*)d_ws;
  size_t off = 0;
  auto carve = [&](size_t bytes) -> void* {
    off = (off + 255) & ~(size_t)255;
    void* p = ws + off;
    off += bytes;
    return p;
  };
  float* deg  = (float*)carve((size_t)N * 4);
  float* dinv = (float*)carve((size_t)N * 4);
  float* agg1 = (float*)carve((size_t)N * DD * 4);
  bf16*  s1   = (bf16*) carve((size_t)TT * N * HH * 2);
  float* agg2 = (float*)carve((size_t)TT * N * HH * 4);
  bf16*  s2   = (bf16*) carve((size_t)TT * N * HH * 2);
  bf16*  W1t  = (bf16*) carve((size_t)HH * DD * 2);
  bf16*  W2t  = (bf16*) carve((size_t)HH * HH * 2);
  bf16*  fcWt = (bf16*) carve((size_t)CP * HH * 2);

  const int B = 256;
  auto blks = [](long long n, int b) { return (unsigned)((n + b - 1) / b); };

  // normalization
  k_deg_init <<<blks(N, B), B, 0, stream>>>(deg, N);
  k_deg_count<<<blks(E, B), B, 0, stream>>>(ei, deg, E);
  k_dinv     <<<blks(N, B), B, 0, stream>>>(deg, dinv, N);

  // weight prep (bf16, transposed, padded)
  k_prep_w<<<blks((long long)HH * DD, B), B, 0, stream>>>(W1,  W1t,  DD, HH, HH * DD);
  k_prep_w<<<blks((long long)HH * HH, B), B, 0, stream>>>(W2,  W2t,  HH, HH, HH * HH);
  k_prep_w<<<blks((long long)CP * HH, B), B, 0, stream>>>(fcW, fcWt, HH, CC, CP * HH);

  // layer 1: aggregate (t-invariant) -> WMMA GEMM + LIF
  k_init_agg1<<<blks((long long)N * DD, B), B, 0, stream>>>(feat, dinv, agg1, N);
  k_edge_agg1<<<blks((long long)E * 32, B), B, 0, stream>>>(ei, feat, dinv, agg1, E);
  k_gemm1_lif<<<blks((long long)(N >> 4) * 4 * 32, B), B, 0, stream>>>(agg1, W1t, b1, s1, N);

  // layer 2: aggregate spikes per t -> WMMA GEMM + LIF
  k_init_agg2<<<blks((long long)TT * N * HH, B), B, 0, stream>>>(s1, dinv, agg2, N, TT * N * HH);
  k_edge_agg2<<<blks((long long)E * TT * 32, B), B, 0, stream>>>(ei, s1, dinv, agg2, E, N);
  k_gemm2_lif<<<blks((long long)(N >> 4) * 4 * 32, B), B, 0, stream>>>(agg2, W2t, b2, s2, N);

  // FC + LIF + log-mean output (one wave per 16-row tile, all 48 padded cols)
  k_fc_lif_out<<<blks((long long)(N >> 4) * 32, B), B, 0, stream>>>(s2, fcWt, fcb, out, N);
}
